// Transformer_12214886990303
// MI455X (gfx1250) — matrix operand order repaired and checked by reading, not compiled, not run on hardware
//
#include <hip/hip_runtime.h>
#include <hip/hip_fp16.h>

typedef __attribute__((ext_vector_type(16))) _Float16 v16h;
typedef __attribute__((ext_vector_type(8)))  _Float16 v8h;
typedef __attribute__((ext_vector_type(8)))  float    v8f;

#define NSEQ 2304
#define CDIM 256
#define NHEADS 8
#define HDIM 32
#define HIDDEN 1024
#define QKVDIM 768
#define QKSCALE 0.17677669529663687f   // 32^-0.5
#define LOG2E 1.44269504088896340736f

// ---------------------------------------------------------------------------
// WMMA fragment helpers (wave32, v_wmma_f32_16x16x32_f16)
// A 16x32 f16: lane<16 -> row=lane, K = [kk..kk+15]
//              lane>=16 -> row=lane-16, K = [kk+16..kk+31], 16 contiguous per lane
// B 32x16 f16 (from B^T row-major): lane&15 -> col, lane<16 K=[kk..kk+15],
//              lane>=16 K=[kk+16..kk+31], 16 contiguous per lane.
// C/D 16x16 f32: VGPR r -> M=r (lanes 0-15) / M=r+8 (lanes 16-31), N = lane&15
// ---------------------------------------------------------------------------
static __device__ __forceinline__ v16h make16(v8h lo, v8h hi) {
  return __builtin_shufflevector(lo, hi, 0,1,2,3,4,5,6,7,8,9,10,11,12,13,14,15);
}

static __device__ __forceinline__ v16h load_a_frag(const _Float16* __restrict__ A,
                                                   int lda, int m0, int kk, int lane) {
  int r  = lane & 15;
  int k0 = kk + ((lane & 16) ? 16 : 0);
  const _Float16* p = A + (size_t)(m0 + r) * lda + k0;
  v8h lo = *(const v8h*)(p);
  v8h hi = *(const v8h*)(p + 8);
  return make16(lo, hi);
}

static __device__ __forceinline__ v16h load_b_frag(const _Float16* __restrict__ BT,
                                                   int ldb, int n0, int kk, int lane) {
  int c  = lane & 15;
  int k0 = kk + ((lane & 16) ? 16 : 0);
  const _Float16* p = BT + (size_t)(n0 + c) * ldb + k0;
  v8h lo = *(const v8h*)(p);
  v8h hi = *(const v8h*)(p + 8);
  return make16(lo, hi);
}

static __device__ __forceinline__ v8f wmma16(v16h a, v16h b, v8f c) {
  return __builtin_amdgcn_wmma_f32_16x16x32_f16(false, a, false, b, (short)0, c,
                                                false, false);
}

// ---------------------------------------------------------------------------
// Generic WMMA GEMM: C[M,Nn] = A[M,K] * B^T[Nn,K], epilogue functor per element
// Block = 256 threads = 8 waves; WG tile 128x128, wave tile 32x64
// (8 WMMA accumulators, 6 fragment loads per 32-wide K step)
// ---------------------------------------------------------------------------
template <class Epi>
__global__ __launch_bounds__(256) void gemm_wmma(const _Float16* __restrict__ A,
                                                 const _Float16* __restrict__ BT,
                                                 int Nn, int K, Epi epi) {
  const int tiles_n = Nn >> 7;
  const int bm = (blockIdx.x / tiles_n) << 7;
  const int bn = (blockIdx.x % tiles_n) << 7;
  const int wave = threadIdx.x >> 5;
  const int lane = threadIdx.x & 31;
  const int wm = bm + (wave >> 1) * 32;
  const int wn = bn + (wave & 1) * 64;

  v8f acc0[4] = {{}, {}, {}, {}};
  v8f acc1[4] = {{}, {}, {}, {}};
  for (int kk = 0; kk < K; kk += 32) {
    if (kk + 64 <= K) {  // keep next K-slab warm (everything is L2-resident)
      __builtin_prefetch(A + (size_t)(wm + (lane & 15)) * K + kk + 32, 0, 1);
      __builtin_prefetch(BT + (size_t)(wn + (lane & 15)) * K + kk + 32, 0, 1);
      __builtin_prefetch(BT + (size_t)(wn + 32 + (lane & 15)) * K + kk + 32, 0, 1);
    }
    v16h a0 = load_a_frag(A, K, wm, kk, lane);
    v16h a1 = load_a_frag(A, K, wm + 16, kk, lane);
#pragma unroll
    for (int j = 0; j < 4; ++j) {
      v16h b = load_b_frag(BT, K, wn + j * 16, kk, lane);
      acc0[j] = wmma16(a0, b, acc0[j]);
      acc1[j] = wmma16(a1, b, acc1[j]);
    }
  }

  const int hrow = (lane & 16) ? 8 : 0;
  const int col  = lane & 15;
#pragma unroll
  for (int j = 0; j < 4; ++j)
#pragma unroll
    for (int r = 0; r < 8; ++r) {
      epi(wm + hrow + r,      wn + j * 16 + col, acc0[j][r]);
      epi(wm + 16 + hrow + r, wn + j * 16 + col, acc1[j][r]);
    }
}

// ---------------------------- epilogue functors ----------------------------
struct EpiStoreF16 {  // plain f16 store (used by the QKV GEMM)
  _Float16* out; int ldc;
  __device__ void operator()(int n, int c, float v) const {
    out[(size_t)n * ldc + c] = (_Float16)v;
  }
};
struct EpiBiasResF32 {  // out = acc + bias[c] + res (fp32)
  const float *bias, *res; float* out;
  __device__ void operator()(int n, int c, float v) const {
    size_t i = (size_t)n * CDIM + c;
    out[i] = v + bias[c] + res[i];
  }
};
struct EpiGeluF16 {  // out = gelu_exact(acc + bias[c]) as f16
  const float* bias; _Float16* out;
  __device__ void operator()(int n, int c, float v) const {
    float u = v + bias[c];
    float g = 0.5f * u * (1.0f + erff(u * 0.70710678118654752f));
    out[(size_t)n * HIDDEN + c] = (_Float16)g;
  }
};

// V slice of qkv [N,768] -> Vt [NHEADS*HDIM][NSEQ] (coalesced read, f16)
__global__ __launch_bounds__(256) void vt_build(const _Float16* __restrict__ qkv,
                                                _Float16* __restrict__ Vt) {
  int i = blockIdx.x * 256 + threadIdx.x;   // over NSEQ*CDIM
  int n = i >> 8, hd = i & 255;
  Vt[(size_t)hd * NSEQ + n] = qkv[(size_t)n * QKVDIM + 2 * CDIM + hd];
}

// ---------------------------------------------------------------------------
// Flash attention: 1 wave = 16 query rows of one head; Q/K read directly from
// the qkv buffer (head slice = base + h*32, ld = 768); 1/sqrt(d) folded into
// the Q fragment. 64-column steps amortize the softmax shuffle overhead over
// 10 WMMAs/step (4 QK^T + 2 ones-rowsum + 4 P*V).
//  * one shared running max per wave-tile (exact: softmax is invariant to any
//    per-row-consistent shift; only an underflow hazard)
//  * row sums via P x ones WMMA -> lands in the accumulator layout, zero
//    shuffles, bit-consistent with the f16 P used for P*V
//  * bias gather: a tile's 16 rows share one image row (48 = 3*16), so the
//    analytic rel-pos index is (base + r): one pointer, 8 imm-offset loads
// Grid: 144 blocks * 8 waves = 8 heads * 144 row-tiles.
// ---------------------------------------------------------------------------
__global__ __launch_bounds__(256) void attn_flash(
    const _Float16* __restrict__ qkv, const _Float16* __restrict__ Vt,
    const float* __restrict__ bias_table, _Float16* __restrict__ attn_out) {
  __shared__ alignas(16) _Float16 Pbuf[8][16 * 64];
  const int wave = threadIdx.x >> 5;
  const int lane = threadIdx.x & 31;
  const int task = blockIdx.x * 8 + wave;
  const int h  = task / 144;
  const int m0 = (task % 144) * 16;

  const _Float16* Qb = qkv + h * HDIM;            // [NSEQ, 32] with ld = 768
  const _Float16* Kb = qkv + CDIM + h * HDIM;     // [NSEQ, 32] with ld = 768
  const _Float16* V  = Vt + (size_t)h * HDIM * NSEQ;

  v16h aQ = load_a_frag(Qb, QKVDIM, m0, 0, lane);  // full head dim, one frag
#pragma unroll
  for (int i = 0; i < 16; ++i) aQ[i] = aQ[i] * (_Float16)QKSCALE;

  v16h vones;
#pragma unroll
  for (int i = 0; i < 16; ++i) vones[i] = (_Float16)1.0f;

  const int hrow = (lane & 16) ? 8 : 0;
  const int mi  = m0 / 48;             // constant across the 16-row tile
  const int mjb = (m0 % 48) + hrow;    // mj for r=0; mj(r) = mjb + r

  float mrun = -3.0e30f;
  v8f lacc = {}, o0 = {}, o1 = {};

  for (int j0 = 0; j0 < NSEQ; j0 += 64) {
    // ---- scores: four 16-col tiles, one WMMA each (K-dim == HDIM == 32) ----
    v8f s[4];
#pragma unroll
    for (int t = 0; t < 4; ++t) {
      v16h bK = load_b_frag(Kb, QKVDIM, j0 + t * 16, 0, lane);
      s[t] = wmma16(aQ, bK, v8f{});
    }

    // ---- relative-position bias: analytic index, imm-offset loads ----
#pragma unroll
    for (int t = 0; t < 4; ++t) {
      int n  = j0 + t * 16 + (lane & 15);
      int ni = n / 48, nj = n - ni * 48;
      const float* bp =
          bias_table + (size_t)((mi - ni + 47) * 95 + (mjb - nj + 47)) * NHEADS + h;
#pragma unroll
      for (int r = 0; r < 8; ++r) s[t][r] += bp[r * NHEADS];
    }

    // ---- shared running max for the whole tile (exact; see note above) ----
    float tm = s[0][0];
#pragma unroll
    for (int t = 0; t < 4; ++t)
#pragma unroll
      for (int r = (t == 0) ? 1 : 0; r < 8; ++r) tm = fmaxf(tm, s[t][r]);
#pragma unroll
    for (int msk = 1; msk < 32; msk <<= 1) tm = fmaxf(tm, __shfl_xor(tm, msk, 32));
    float nm  = fmaxf(mrun, tm);
    float rsc = exp2f((mrun - nm) * LOG2E);
    mrun = nm;

    // ---- P = exp(s - nm), straight into LDS (C-layout -> row-major) ----
    _Float16* P = Pbuf[wave];
    const int cc = lane & 15;
#pragma unroll
    for (int t = 0; t < 4; ++t)
#pragma unroll
      for (int r = 0; r < 8; ++r)
        P[(hrow + r) * 64 + t * 16 + cc] = (_Float16)exp2f((s[t][r] - nm) * LOG2E);
#pragma unroll
    for (int r = 0; r < 8; ++r) { o0[r] *= rsc; o1[r] *= rsc; lacc[r] *= rsc; }
    __syncthreads();  // uniform trip count across all 8 waves

    // ---- P is 16x64: two A fragments; row sums via ones-WMMA; P*V ----
    v16h aP0 = load_a_frag(P, 64, 0, 0, lane);
    v16h aP1 = load_a_frag(P, 64, 0, 32, lane);
    lacc = wmma16(aP0, vones, lacc);
    lacc = wmma16(aP1, vones, lacc);
    v16h bV00 = load_b_frag(V, NSEQ, 0,  j0, lane);
    v16h bV10 = load_b_frag(V, NSEQ, 16, j0, lane);
    o0 = wmma16(aP0, bV00, o0);
    o1 = wmma16(aP0, bV10, o1);
    v16h bV01 = load_b_frag(V, NSEQ, 0,  j0 + 32, lane);
    v16h bV11 = load_b_frag(V, NSEQ, 16, j0 + 32, lane);
    o0 = wmma16(aP1, bV01, o0);
    o1 = wmma16(aP1, bV11, o1);
  }

  // ---- normalize and write [N, NHEADS*HDIM] f16 ----
#pragma unroll
  for (int r = 0; r < 8; ++r) {
    float inv = 1.0f / lacc[r];
    size_t row = (size_t)(m0 + hrow + r) * CDIM + h * HDIM;
    attn_out[row + (lane & 15)]      = (_Float16)(o0[r] * inv);
    attn_out[row + 16 + (lane & 15)] = (_Float16)(o1[r] * inv);
  }
}

// ---------------------------------------------------------------------------
// LayerNorm over C=256, one row per block (256 threads), f16 output
// ---------------------------------------------------------------------------
__global__ __launch_bounds__(256) void layernorm_f16(const float* __restrict__ x,
                                                     const float* __restrict__ g,
                                                     const float* __restrict__ b,
                                                     _Float16* __restrict__ out) {
  __shared__ float sp[8];
  const int row = blockIdx.x, tid = threadIdx.x;
  float v = x[(size_t)row * CDIM + tid];

  float s = v;
#pragma unroll
  for (int m = 16; m >= 1; m >>= 1) s += __shfl_xor(s, m, 32);
  if ((tid & 31) == 0) sp[tid >> 5] = s;
  __syncthreads();
  float tot = 0.0f;
#pragma unroll
  for (int i = 0; i < 8; ++i) tot += sp[i];
  float mu = tot * (1.0f / CDIM);
  __syncthreads();

  float d = v - mu;
  float s2 = d * d;
#pragma unroll
  for (int m = 16; m >= 1; m >>= 1) s2 += __shfl_xor(s2, m, 32);
  if ((tid & 31) == 0) sp[tid >> 5] = s2;
  __syncthreads();
  float tv = 0.0f;
#pragma unroll
  for (int i = 0; i < 8; ++i) tv += sp[i];
  float rstd = rsqrtf(tv * (1.0f / CDIM) + 1e-5f);
  out[(size_t)row * CDIM + tid] = (_Float16)(d * rstd * g[tid] + b[tid]);
}

// fp32 weight [K,N] -> fp16 transposed [N,K]
__global__ __launch_bounds__(256) void wcvt_t(const float* __restrict__ W,
                                              _Float16* __restrict__ WT,
                                              int K, int N) {
  int i = blockIdx.x * 256 + threadIdx.x;
  if (i >= K * N) return;
  int k = i / N, n = i - k * N;
  WT[(size_t)n * K + k] = (_Float16)W[i];
}

// ---------------------------------------------------------------------------
extern "C" void kernel_launch(void* const* d_in, const int* in_sizes, int n_in,
                              void* d_out, int out_size, void* d_ws, size_t ws_size,
                              hipStream_t stream) {
  const float* x          = (const float*)d_in[0];
  const float* gamma1     = (const float*)d_in[1];
  const float* beta1      = (const float*)d_in[2];
  const float* w_qkv      = (const float*)d_in[3];
  const float* w_proj     = (const float*)d_in[4];
  const float* b_proj     = (const float*)d_in[5];
  const float* bias_table = (const float*)d_in[6];
  const float* gamma2     = (const float*)d_in[7];
  const float* beta2      = (const float*)d_in[8];
  const float* w_fc1      = (const float*)d_in[9];
  const float* b_fc1      = (const float*)d_in[10];
  const float* w_fc2      = (const float*)d_in[11];
  const float* b_fc2      = (const float*)d_in[12];
  // d_in[13] = rel_idx (unused: index computed analytically on device)
  float* out = (float*)d_out;

  // workspace carve-up (all offsets 16B-aligned)
  char* ws = (char*)d_ws;
  _Float16* wqkvT   = (_Float16*)ws;                 ws += (size_t)QKVDIM * CDIM * 2;
  _Float16* wprojT  = (_Float16*)ws;                 ws += (size_t)CDIM * CDIM * 2;
  _Float16* wfc1T   = (_Float16*)ws;                 ws += (size_t)HIDDEN * CDIM * 2;
  _Float16* wfc2T   = (_Float16*)ws;                 ws += (size_t)CDIM * HIDDEN * 2;
  _Float16* t1      = (_Float16*)ws;                 ws += (size_t)NSEQ * CDIM * 2;
  _Float16* qkvbuf  = (_Float16*)ws;                 ws += (size_t)NSEQ * QKVDIM * 2;
  _Float16* Vt      = (_Float16*)ws;                 ws += (size_t)NSEQ * CDIM * 2;
  _Float16* attn_o  = (_Float16*)ws;                 ws += (size_t)NSEQ * CDIM * 2;
  float*    y1      = (float*)ws;                    ws += (size_t)NSEQ * CDIM * 4;
  _Float16* t2      = (_Float16*)ws;                 ws += (size_t)NSEQ * CDIM * 2;
  _Float16* hbuf    = (_Float16*)ws;                 ws += (size_t)NSEQ * HIDDEN * 2;

  // 1) weights -> f16 transposed
  wcvt_t<<<(QKVDIM * CDIM + 255) / 256, 256, 0, stream>>>(w_qkv, wqkvT, CDIM, QKVDIM);
  wcvt_t<<<(CDIM * CDIM + 255) / 256, 256, 0, stream>>>(w_proj, wprojT, CDIM, CDIM);
  wcvt_t<<<(CDIM * HIDDEN + 255) / 256, 256, 0, stream>>>(w_fc1, wfc1T, CDIM, HIDDEN);
  wcvt_t<<<(HIDDEN * CDIM + 255) / 256, 256, 0, stream>>>(w_fc2, wfc2T, HIDDEN, CDIM);

  // 2) LN1
  layernorm_f16<<<NSEQ, 256, 0, stream>>>(x, gamma1, beta1, t1);

  // 3) QKV GEMM [2304,256]x[256,768] -> qkvbuf; then V slice -> Vt
  gemm_wmma<<<(NSEQ / 128) * (QKVDIM / 128), 256, 0, stream>>>(
      t1, wqkvT, QKVDIM, CDIM, EpiStoreF16{qkvbuf, QKVDIM});
  vt_build<<<(NSEQ * CDIM) / 256, 256, 0, stream>>>(qkvbuf, Vt);

  // 4) flash attention (8 heads x 144 row-tiles = 1152 wave tasks)
  attn_flash<<<144, 256, 0, stream>>>(qkvbuf, Vt, bias_table, attn_o);

  // 5) proj GEMM + bias + residual -> y1 (fp32)
  gemm_wmma<<<(NSEQ / 128) * (CDIM / 128), 256, 0, stream>>>(
      attn_o, wprojT, CDIM, CDIM, EpiBiasResF32{b_proj, x, y1});

  // 6) LN2
  layernorm_f16<<<NSEQ, 256, 0, stream>>>(y1, gamma2, beta2, t2);

  // 7) FC1 + exact GELU -> hbuf (f16)
  gemm_wmma<<<(NSEQ / 128) * (HIDDEN / 128), 256, 0, stream>>>(
      t2, wfc1T, HIDDEN, CDIM, EpiGeluF16{b_fc1, hbuf});

  // 8) FC2 + bias + residual -> out (fp32)
  gemm_wmma<<<(NSEQ / 128) * (CDIM / 128), 256, 0, stream>>>(
      hbuf, wfc2T, CDIM, HIDDEN, EpiBiasResF32{b_fc2, y1, out});
}
